// SingleStreamBlock_81741817577931
// MI455X (gfx1250) — matrix operand order repaired and checked by reading, not compiled, not run on hardware
//
#include <hip/hip_runtime.h>
#include <hip/hip_bf16.h>
#include <math.h>

typedef __bf16 bf16;
typedef __attribute__((ext_vector_type(16))) __bf16 v16bf;
typedef __attribute__((ext_vector_type(8)))  __bf16 v8bf;
typedef __attribute__((ext_vector_type(4)))  __bf16 v4bf;
typedef __attribute__((ext_vector_type(8)))  float  v8f;
typedef __attribute__((ext_vector_type(4)))  unsigned int u32x4;
typedef __attribute__((ext_vector_type(8)))  unsigned int u32x8;

#define L_    2048
#define D_    3072
#define H_    24
#define HD_   128
#define MLP_  12288
#define N1_   21504   /* 3*D + MLP */
#define NCAT_ 15360   /* D + MLP   */
#define NQKV_ 9216    /* 3*D       */

// ---------------------------------------------------------------------------
// WMMA helpers (CDNA5 wave32: 16x16x32 bf16, f32 accumulate)
// ---------------------------------------------------------------------------
__device__ __forceinline__ v8f wmma_bf16(v16bf a, v16bf b, v8f c) {
  return __builtin_amdgcn_wmma_f32_16x16x32_bf16(false, a, false, b, (short)0, c,
                                                 false, false);
}

__device__ __forceinline__ v16bf cat8(v8bf lo, v8bf hi) {
  return __builtin_shufflevector(lo, hi, 0,1,2,3,4,5,6,7,8,9,10,11,12,13,14,15);
}

// A fragment: 16x32 bf16 tile, row-major (rows = M, K contiguous), ld = elems.
__device__ __forceinline__ v16bf ld_frag_a(const bf16* p, int ld) {
  int lane = threadIdx.x & 31;
  const bf16* row = p + (lane & 15) * ld + ((lane >> 4) << 3);
  v8bf lo = *(const v8bf*)(row);
  v8bf hi = *(const v8bf*)(row + 16);
  return cat8(lo, hi);
}

// B fragment for a K-major operand (tile rows = N, K contiguous), ld = elems.
__device__ __forceinline__ v16bf ld_frag_bt(const bf16* p, int ld) {
  int lane = threadIdx.x & 31;
  const bf16* row = p + (lane & 15) * ld + ((lane >> 4) << 3);
  v8bf lo = *(const v8bf*)(row);
  v8bf hi = *(const v8bf*)(row + 16);
  return cat8(lo, hi);
}

// ---------------------------------------------------------------------------
// CDNA5 async / TDM data movement
// ---------------------------------------------------------------------------
// Per-lane async global -> LDS, 16B, tracked by ASYNCcnt (no VGPR staging).
__device__ __forceinline__ void async_ld_b128(const void* gptr, void* lptr) {
  unsigned lds = (unsigned)(size_t)lptr;           // LDS offset = low 32 bits
  unsigned long long ga = (unsigned long long)(size_t)gptr;
  asm volatile("global_load_async_to_lds_b128 %0, %1, off"
               :: "v"(lds), "v"(ga) : "memory");
}

__device__ __forceinline__ void wait_async0() {
  asm volatile("s_wait_asynccnt 0x0" ::: "memory");
}

// Tensor Data Mover: contiguous 1-D copy of `elems` bf16 global -> LDS.
// Builds D# group0 (4 SGPRs) + group1 (8 SGPRs), issues tensor_load_to_lds.
// Tracked by TENSORcnt. Issue from a single wave; EXEC is ignored by TDM.
__device__ __forceinline__ void tdm_copy_bf16(const void* gsrc, void* ldst,
                                              unsigned elems) {
  unsigned long long ga = (unsigned long long)(size_t)gsrc;
  unsigned lds = (unsigned)(size_t)ldst;
  u32x4 g0;
  g0[0] = 1u;                                       // count=1, user D#
  g0[1] = lds;                                      // lds_addr (bytes)
  g0[2] = (unsigned)(ga & 0xffffffffu);             // global_addr[31:0]
  g0[3] = (unsigned)((ga >> 32) & 0x01ffffffu)      // global_addr[56:32]
        | (2u << 30);                               // type=2 ("image")
  u32x8 g1;
  g1[0] = (1u << 16);                               // data_size=1 -> 2 bytes
  g1[1] = (elems & 0xffffu) << 16;                  // tensor_dim0[15:0]
  g1[2] = (elems >> 16) | (1u << 16);               // tensor_dim0[31:16], tensor_dim1=1
  g1[3] = (elems & 0xffffu) << 16;                  // tile_dim0 = elems
  g1[4] = 1u;                                       // tile_dim1=1, tile_dim2=0
  g1[5] = elems;                                    // tensor_dim0_stride[31:0]
  g1[6] = 0u;                                       // stride hi / dim1_stride lo
  g1[7] = 0u;
  asm volatile("tensor_load_to_lds %0, %1" :: "s"(g0), "s"(g1) : "memory");
}

// ---------------------------------------------------------------------------
// Kernel 1: m = silu(vec) @ mod_w^T + mod_b   (one wave per output row)
// ---------------------------------------------------------------------------
__global__ __launch_bounds__(256) void mod_gemv_kernel(
    const float* __restrict__ vec, const float* __restrict__ mw,
    const float* __restrict__ mb, float* __restrict__ m, int N, int K)
{
  int wave = blockIdx.x * 8 + (threadIdx.x >> 5);
  int lane = threadIdx.x & 31;
  if (wave >= N) return;
  const float* row = mw + (size_t)wave * K;
  float sum = 0.f;
  for (int k = lane; k < K; k += 32) {
    float a = vec[k];
    float s = a / (1.f + __expf(-a));
    sum += s * row[k];
  }
  #pragma unroll
  for (int off = 16; off; off >>= 1) sum += __shfl_down(sum, off, 32);
  if (lane == 0) m[wave] = sum + mb[wave];
}

// ---------------------------------------------------------------------------
// Kernel 2: f32 -> bf16 conversion (weights)
// ---------------------------------------------------------------------------
__global__ __launch_bounds__(256) void f32_to_bf16_kernel(
    const float* __restrict__ in, bf16* __restrict__ out, size_t n)
{
  size_t i = ((size_t)blockIdx.x * blockDim.x + threadIdx.x) * 4;
  if (i + 3 < n) {
    float4 v = *(const float4*)(in + i);
    v4bf o; o.x = (bf16)v.x; o.y = (bf16)v.y; o.z = (bf16)v.z; o.w = (bf16)v.w;
    *(v4bf*)(out + i) = o;
  } else {
    for (; i < n; i++) out[i] = (bf16)in[i];
  }
}

// ---------------------------------------------------------------------------
// Kernel 3: LayerNorm + modulation -> bf16 x_mod
// ---------------------------------------------------------------------------
__global__ __launch_bounds__(256) void ln_mod_kernel(
    const float* __restrict__ x, const float* __restrict__ mshift,
    const float* __restrict__ mscale, bf16* __restrict__ xmod)
{
  int row = blockIdx.x;
  const float* xr = x + (size_t)row * D_;
  float s = 0.f, s2 = 0.f;
  for (int d = threadIdx.x; d < D_; d += 256) {
    float v = xr[d]; s += v; s2 += v * v;
  }
  __shared__ float rb[2][8];
  int lane = threadIdx.x & 31, wave = threadIdx.x >> 5;
  #pragma unroll
  for (int off = 16; off; off >>= 1) {
    s  += __shfl_down(s,  off, 32);
    s2 += __shfl_down(s2, off, 32);
  }
  if (lane == 0) { rb[0][wave] = s; rb[1][wave] = s2; }
  __syncthreads();
  s = 0.f; s2 = 0.f;
  #pragma unroll
  for (int w = 0; w < 8; w++) { s += rb[0][w]; s2 += rb[1][w]; }
  float mu   = s  * (1.f / D_);
  float var  = s2 * (1.f / D_) - mu * mu;
  float rstd = rsqrtf(var + 1e-6f);
  bf16* orow = xmod + (size_t)row * D_;
  for (int d = threadIdx.x; d < D_; d += 256) {
    float ln = (xr[d] - mu) * rstd;
    orow[d] = (bf16)((1.f + mscale[d]) * ln + mshift[d]);
  }
}

// ---------------------------------------------------------------------------
// Kernel 4: C[M,N] = A[M,K] @ W[N,K]^T   (both K-major, bf16, f32 accum)
// 128x128 block tile, K-step 32, double-buffered LDS filled by
// global_load_async_to_lds_b128 (ASYNCcnt) overlapping the WMMAs.
// mode 0: Out = acc + bias                (lin1 -> proj, f32)
// mode 1: Out = resid + gate*(acc+bias)   (lin2 -> final output)
// ---------------------------------------------------------------------------
__global__ __launch_bounds__(256) void gemm_bf16_nt(
    const bf16* __restrict__ A, const bf16* __restrict__ W,
    float* __restrict__ Out, const float* __restrict__ bias,
    const float* __restrict__ resid, const float* __restrict__ gate,
    int M, int N, int K, int mode)
{
  __shared__ __align__(16) bf16 sA[2][128 * 32];
  __shared__ __align__(16) bf16 sB[2][128 * 32];
  int n0 = blockIdx.x * 128;
  int m0 = blockIdx.y * 128;
  int wave = threadIdx.x >> 5;
  int lane = threadIdx.x & 31;
  int wm = (wave & 1) * 64;
  int wn = (wave >> 1) * 32;

  v8f z8 = {0.f,0.f,0.f,0.f,0.f,0.f,0.f,0.f};
  v8f acc[4][2];
  #pragma unroll
  for (int i = 0; i < 4; i++)
    #pragma unroll
    for (int j = 0; j < 2; j++) acc[i][j] = z8;

  int r0 = threadIdx.x >> 2;           // 0..63
  int r1 = r0 + 64;                    // 64..127
  int cc = (threadIdx.x & 3) << 3;     // 0,8,16,24

  auto stage = [&](int buf, int k0) {
    async_ld_b128(A + (size_t)(m0 + r0) * K + k0 + cc, &sA[buf][r0 * 32 + cc]);
    async_ld_b128(A + (size_t)(m0 + r1) * K + k0 + cc, &sA[buf][r1 * 32 + cc]);
    async_ld_b128(W + (size_t)(n0 + r0) * K + k0 + cc, &sB[buf][r0 * 32 + cc]);
    async_ld_b128(W + (size_t)(n0 + r1) * K + k0 + cc, &sB[buf][r1 * 32 + cc]);
  };

  stage(0, 0);
  wait_async0();
  __syncthreads();

  int cur = 0;
  for (int k0 = 0; k0 < K; k0 += 32) {
    if (k0 + 32 < K) {
      stage(cur ^ 1, k0 + 32);                       // DMA next tile
      __builtin_prefetch(A + (size_t)(m0 + (threadIdx.x >> 1)) * K + k0 + 64, 0, 0);
      __builtin_prefetch(W + (size_t)(n0 + (threadIdx.x >> 1)) * K + k0 + 64, 0, 0);
    }
    v16bf bf0 = ld_frag_bt(&sB[cur][(wn +  0) * 32], 32);
    v16bf bf1 = ld_frag_bt(&sB[cur][(wn + 16) * 32], 32);
    #pragma unroll
    for (int i = 0; i < 4; i++) {
      v16bf af = ld_frag_a(&sA[cur][(wm + 16 * i) * 32], 32);
      acc[i][0] = wmma_bf16(af, bf0, acc[i][0]);
      acc[i][1] = wmma_bf16(af, bf1, acc[i][1]);
    }
    wait_async0();                                   // next tile landed
    __syncthreads();
    cur ^= 1;
  }

  int hf = lane >> 4, nc = lane & 15;
  #pragma unroll
  for (int i = 0; i < 4; i++)
    #pragma unroll
    for (int j = 0; j < 2; j++)
      #pragma unroll
      for (int r = 0; r < 8; r++) {
        int row = m0 + wm + 16 * i + r + 8 * hf;
        int col = n0 + wn + 16 * j + nc;
        float v = acc[i][j][r] + bias[col];
        if (mode == 0) {
          Out[(size_t)row * N + col] = v;
        } else {
          Out[(size_t)row * N + col] =
              resid[(size_t)row * N + col] + gate[col] * v;
        }
      }
}

// ---------------------------------------------------------------------------
// Kernel 5: RMS-norm(q,k) + RoPE + pack v  -> [H][L][HD] bf16
// ---------------------------------------------------------------------------
__global__ __launch_bounds__(128) void qkv_prep_kernel(
    const float* __restrict__ proj, const float* __restrict__ pe,
    const float* __restrict__ qs, const float* __restrict__ ks,
    bf16* __restrict__ Qh, bf16* __restrict__ Kh, bf16* __restrict__ Vh)
{
  int l = blockIdx.x, h = blockIdx.y, d = threadIdx.x;
  const float* base = proj + (size_t)l * N1_;
  float q = base[h * HD_ + d];
  float k = base[D_ + h * HD_ + d];
  float v = base[2 * D_ + h * HD_ + d];

  __shared__ float rb[2][4];
  __shared__ float sq[128], sk[128];
  float qq = q * q, kk = k * k;
  int lane = d & 31, wave = d >> 5;
  #pragma unroll
  for (int off = 16; off; off >>= 1) {
    qq += __shfl_down(qq, off, 32);
    kk += __shfl_down(kk, off, 32);
  }
  if (lane == 0) { rb[0][wave] = qq; rb[1][wave] = kk; }
  __syncthreads();
  float qss = 0.f, kss = 0.f;
  #pragma unroll
  for (int w = 0; w < 4; w++) { qss += rb[0][w]; kss += rb[1][w]; }
  float qn = q * rsqrtf(qss * (1.f / HD_) + 1e-6f) * qs[d];
  float kn = k * rsqrtf(kss * (1.f / HD_) + 1e-6f) * ks[d];
  sq[d] = qn; sk[d] = kn;
  __syncthreads();

  int i = d >> 1, j = d & 1;
  const float* per = pe + (((size_t)l * 64 + i) * 2 + j) * 2;  // pe[l][i][j][:]
  float qo = per[0] * sq[2 * i] + per[1] * sq[2 * i + 1];
  float ko = per[0] * sk[2 * i] + per[1] * sk[2 * i + 1];
  size_t o = ((size_t)h * L_ + l) * HD_ + d;
  Qh[o] = (bf16)qo; Kh[o] = (bf16)ko; Vh[o] = (bf16)v;
}

// ---------------------------------------------------------------------------
// Kernel 6: flash attention, one (head, 128-query block) per workgroup.
// Q and K tiles are contiguous 32KB blocks -> Tensor Data Mover loads
// (wave 0 issues, TENSORcnt wait, barrier). V is vector-loaded + transposed.
// LDS: Q 32KB + K 32KB + V^T 32KB + P 32KB = 128KB dynamic (WGP has 320KB).
// ---------------------------------------------------------------------------
__global__ __launch_bounds__(256) void flash_attn_kernel(
    const bf16* __restrict__ Q, const bf16* __restrict__ Kh,
    const bf16* __restrict__ V, bf16* __restrict__ Cat)
{
  extern __shared__ __align__(16) bf16 smem[];
  bf16* sQ = smem;                 // [128][128] rows=q, d contiguous
  bf16* sK = smem + 128 * 128;     // [128][128] rows=key, d contiguous
  bf16* sV = smem + 2 * 128 * 128; // [128][128] rows=d, key contiguous
  bf16* sP = smem + 3 * 128 * 128; // 8 waves * [16][128]

  int h = blockIdx.y;
  int q0 = blockIdx.x * 128;
  int wave = threadIdx.x >> 5;
  int lane = threadIdx.x & 31;
  int hf = lane >> 4, nc = lane & 15;
  const size_t hb = (size_t)h * L_ * HD_;
  bf16* sPw = sP + wave * 16 * HD_;
  const float scale = 0.08838834764831845f;  // 1/sqrt(128)

  // Q tile via TDM (wave 0 drives the DMA engine; EXEC ignored by TDM)
  if (threadIdx.x < 32) {
    tdm_copy_bf16(Q + hb + (size_t)q0 * HD_, sQ, 128 * HD_);
    __builtin_amdgcn_s_wait_tensorcnt(0);
  }
  __syncthreads();

  v8f z8 = {0.f,0.f,0.f,0.f,0.f,0.f,0.f,0.f};
  v8f accO[8];
  float mrun[8], lrun[8];
  #pragma unroll
  for (int j = 0; j < 8; j++) accO[j] = z8;
  #pragma unroll
  for (int r = 0; r < 8; r++) { mrun[r] = -1e30f; lrun[r] = 0.f; }

  for (int kb = 0; kb < L_ / 128; kb++) {
    __syncthreads();
    int kbase = kb * 128;
    if (threadIdx.x < 32)                          // K tile via TDM
      tdm_copy_bf16(Kh + hb + (size_t)kbase * HD_, sK, 128 * HD_);
    #pragma unroll
    for (int it = 0; it < 8; it++) {               // V tile, transposed
      int c = threadIdx.x + 256 * it;
      int row = c >> 4, col = (c & 15) << 3;
      v8bf vv = *(const v8bf*)(V + hb + (size_t)(kbase + row) * HD_ + col);
      #pragma unroll
      for (int e = 0; e < 8; e++) sV[(col + e) * 128 + row] = vv[e];
    }
    if (threadIdx.x < 32) __builtin_amdgcn_s_wait_tensorcnt(0);
    __syncthreads();

    // S = Q @ K^T  (both d-major -> symmetric fragments)
    v8f accS[8];
    #pragma unroll
    for (int j = 0; j < 8; j++) accS[j] = z8;
    #pragma unroll
    for (int dk = 0; dk < 4; dk++) {
      v16bf af = ld_frag_a(sQ + (wave * 16) * HD_ + dk * 32, HD_);
      #pragma unroll
      for (int j = 0; j < 8; j++) {
        v16bf bfv = ld_frag_bt(sK + (16 * j) * HD_ + dk * 32, HD_);
        accS[j] = wmma_bf16(af, bfv, accS[j]);
      }
    }

    // Online softmax: rows live on half-waves (C layout: VGPR r -> M=r+8*hf)
    #pragma unroll
    for (int r = 0; r < 8; r++) {
      float mx = mrun[r];
      #pragma unroll
      for (int j = 0; j < 8; j++) mx = fmaxf(mx, accS[j][r] * scale);
      #pragma unroll
      for (int off = 1; off < 16; off <<= 1)
        mx = fmaxf(mx, __shfl_xor(mx, off, 32));
      float alpha = __expf(mrun[r] - mx);
      mrun[r] = mx;
      float rs = 0.f;
      #pragma unroll
      for (int j = 0; j < 8; j++) {
        float p = __expf(accS[j][r] * scale - mx);
        sPw[(r + 8 * hf) * HD_ + 16 * j + nc] = (bf16)p;
        rs += p;
      }
      #pragma unroll
      for (int off = 1; off < 16; off <<= 1) rs += __shfl_xor(rs, off, 32);
      lrun[r] = lrun[r] * alpha + rs;
      #pragma unroll
      for (int j = 0; j < 8; j++) accO[j][r] *= alpha;
    }

    // O += P @ V  (sV is [d][key] so B fragments are contiguous)
    #pragma unroll
    for (int dk = 0; dk < 4; dk++) {
      v16bf af = ld_frag_a(sPw + dk * 32, HD_);
      #pragma unroll
      for (int j = 0; j < 8; j++) {
        v16bf bfv = ld_frag_bt(sV + (16 * j) * 128 + dk * 32, 128);
        accO[j] = wmma_bf16(af, bfv, accO[j]);
      }
    }
  }

  // Normalize and write attention output into cat[:, h*128 + d]
  #pragma unroll
  for (int j = 0; j < 8; j++)
    #pragma unroll
    for (int r = 0; r < 8; r++) {
      int qrow = q0 + wave * 16 + r + 8 * hf;
      int dcol = 16 * j + nc;
      float v = accO[j][r] / lrun[r];
      Cat[(size_t)qrow * NCAT_ + h * HD_ + dcol] = (bf16)v;
    }
}

// ---------------------------------------------------------------------------
// Kernel 7: tanh-GELU on MLP half -> bf16 cat[:, 3072:]
// ---------------------------------------------------------------------------
__global__ __launch_bounds__(256) void gelu_kernel(
    const float* __restrict__ proj, bf16* __restrict__ cat)
{
  size_t idx = (size_t)blockIdx.x * blockDim.x + threadIdx.x;
  if (idx >= (size_t)L_ * MLP_) return;
  int l = (int)(idx / MLP_);
  int j = (int)(idx % MLP_);
  float v = proj[(size_t)l * N1_ + NQKV_ + j];
  float g = 0.5f * v * (1.f + tanhf(0.7978845608f * (v + 0.044715f * v * v * v)));
  cat[(size_t)l * NCAT_ + D_ + j] = (bf16)g;
}

// ---------------------------------------------------------------------------
// Host-side orchestration
// ---------------------------------------------------------------------------
extern "C" void kernel_launch(void* const* d_in, const int* in_sizes, int n_in,
                              void* d_out, int out_size, void* d_ws, size_t ws_size,
                              hipStream_t stream)
{
  (void)in_sizes; (void)n_in; (void)out_size; (void)ws_size;
  const float* x       = (const float*)d_in[0];
  const float* vec     = (const float*)d_in[1];
  const float* pe      = (const float*)d_in[2];
  const float* mod_w   = (const float*)d_in[3];
  const float* mod_b   = (const float*)d_in[4];
  const float* lin1_w  = (const float*)d_in[5];
  const float* lin1_b  = (const float*)d_in[6];
  const float* lin2_w  = (const float*)d_in[7];
  const float* lin2_b  = (const float*)d_in[8];
  const float* q_scale = (const float*)d_in[9];
  const float* k_scale = (const float*)d_in[10];
  float* out = (float*)d_out;

  char* ws = (char*)d_ws;
  size_t off = 0;
  auto take = [&](size_t bytes) -> char* {
    char* p = ws + off;
    off += (bytes + 255) & ~(size_t)255;
    return p;
  };
  float* m_mod = (float*)take((size_t)NQKV_ * sizeof(float));
  bf16*  xmod  = (bf16*) take((size_t)L_ * D_ * 2);
  bf16*  w1    = (bf16*) take((size_t)N1_ * D_ * 2);
  bf16*  w2    = (bf16*) take((size_t)D_ * NCAT_ * 2);
  float* proj  = (float*)take((size_t)L_ * N1_ * sizeof(float));
  bf16*  Qh    = (bf16*) take((size_t)H_ * L_ * HD_ * 2);
  bf16*  Kh    = (bf16*) take((size_t)H_ * L_ * HD_ * 2);
  bf16*  Vh    = (bf16*) take((size_t)H_ * L_ * HD_ * 2);
  bf16*  cat   = (bf16*) take((size_t)L_ * NCAT_ * 2);

  // 1) modulation GEMV
  mod_gemv_kernel<<<NQKV_ / 8, 256, 0, stream>>>(vec, mod_w, mod_b, m_mod,
                                                 NQKV_, D_);
  // 2) weight conversion (bf16 lin1_w = 132MB -> L2-resident during GEMM)
  {
    size_t n1 = (size_t)N1_ * D_;
    f32_to_bf16_kernel<<<(unsigned)((n1 / 4 + 255) / 256), 256, 0, stream>>>(
        lin1_w, w1, n1);
    size_t n2 = (size_t)D_ * NCAT_;
    f32_to_bf16_kernel<<<(unsigned)((n2 / 4 + 255) / 256), 256, 0, stream>>>(
        lin2_w, w2, n2);
  }
  // 3) LN + modulation
  ln_mod_kernel<<<L_, 256, 0, stream>>>(x, m_mod, m_mod + D_, xmod);
  // 4) lin1 GEMM (2048 x 21504 x 3072)
  gemm_bf16_nt<<<dim3(N1_ / 128, L_ / 128), 256, 0, stream>>>(
      xmod, w1, proj, lin1_b, nullptr, nullptr, L_, N1_, D_, 0);
  // 5) q/k RMS + RoPE + v pack
  qkv_prep_kernel<<<dim3(L_, H_), 128, 0, stream>>>(proj, pe, q_scale, k_scale,
                                                    Qh, Kh, Vh);
  // 6) flash attention (24 heads x 16 query blocks), 128KB dynamic LDS
  flash_attn_kernel<<<dim3(L_ / 128, H_), 256, 131072, stream>>>(Qh, Kh, Vh, cat);
  // 7) GELU on MLP half
  {
    size_t n = (size_t)L_ * MLP_;
    gelu_kernel<<<(unsigned)((n + 255) / 256), 256, 0, stream>>>(proj, cat);
  }
  // 8) lin2 GEMM (2048 x 3072 x 15360) with fused bias+gate+residual
  gemm_bf16_nt<<<dim3(D_ / 128, L_ / 128), 256, 0, stream>>>(
      cat, w2, out, lin2_b, x, m_mod + 2 * D_, L_, D_, NCAT_, 1);
}
